// Seq2SeqLSTMGPUv2_53137335386705
// MI455X (gfx1250) — matrix operand order repaired and checked by reading, not compiled, not run on hardware
//
#include <hip/hip_runtime.h>

// ---------------------------------------------------------------------------
// Seq2Seq LSTM + attention for MI455X (gfx1250, wave32, WMMA bf16).
// B=256 S=128 IN=64 OUT=64 H=1024 NH=4 HD=256 STEPS=16
// ---------------------------------------------------------------------------

#define DEVINL __device__ __forceinline__

typedef __bf16 bf16;
typedef __attribute__((ext_vector_type(8)))  __bf16 v8bf;
typedef __attribute__((ext_vector_type(16))) __bf16 v16bf;
typedef __attribute__((ext_vector_type(8)))  float  v8f;
typedef __attribute__((ext_vector_type(4)))  int    v4i;

// Build a 16-element bf16 fragment from two contiguous 16-byte chunks.
DEVINL v16bf load_frag(const bf16* p0, const bf16* p1) {
    v8bf lo = *(const v8bf*)p0;
    v8bf hi = *(const v8bf*)p1;
    v16bf r;
#pragma unroll
    for (int i = 0; i < 8; ++i) { r[i] = lo[i]; r[i + 8] = hi[i]; }
    return r;
}

// Copy 16 bf16 global -> LDS. CDNA5 async-DMA path when available
// (ASYNCcnt-tracked, bypasses VGPRs); plain load+ds_store fallback.
// Builtin signature (from toolchain diagnostics): (v4i AS1*, v4i AS3*, imm, imm).
DEVINL void fill16(const bf16* __restrict__ src, bf16* dst) {
#if defined(__AMDGCN__) && __has_builtin(__builtin_amdgcn_global_load_async_to_lds_b128)
    typedef __attribute__((address_space(1))) v4i gv4i;
    typedef __attribute__((address_space(3))) v4i lv4i;
    __builtin_amdgcn_global_load_async_to_lds_b128((gv4i*)src,       (lv4i*)dst,       0, 0);
    __builtin_amdgcn_global_load_async_to_lds_b128((gv4i*)(src + 8), (lv4i*)(dst + 8), 0, 0);
#else
    *(v8bf*)dst       = *(const v8bf*)src;
    *(v8bf*)(dst + 8) = *(const v8bf*)(src + 8);
#endif
}

DEVINL void wait_fill() {
#if defined(__AMDGCN__) && __has_builtin(__builtin_amdgcn_s_wait_asynccnt)
    __builtin_amdgcn_s_wait_asynccnt(0);
#endif
}

// ---------------------------------------------------------------------------
// Weight prep: fp32 [K,N] (optionally [Kx,N] stacked on [Kh,N]) -> bf16 [N,K].
// ---------------------------------------------------------------------------
__global__ void k_wconv(const float* __restrict__ Wx, int Kx,
                        const float* __restrict__ Wh, int Kh,
                        bf16* __restrict__ Wt, int N) {
    const int Kt = Kx + Kh;
    long long i = (long long)blockIdx.x * blockDim.x + threadIdx.x;
    if (i >= (long long)N * Kt) return;
    int n = (int)(i / Kt);
    int k = (int)(i % Kt);
    float v = (k < Kx) ? Wx[(size_t)k * N + n] : Wh[(size_t)(k - Kx) * N + n];
    Wt[(size_t)n * Kt + k] = (bf16)v;
}

// fp32 (or zero if src==null) -> bf16, strided.
__global__ void k_cvt(const float* __restrict__ src, long long ls,
                      bf16* __restrict__ dst, long long ld,
                      int rows, int cols) {
    int i = blockIdx.x * blockDim.x + threadIdx.x;
    if (i >= rows * cols) return;
    int r = i / cols, c = i % cols;
    float v = src ? src[(size_t)r * ls + c] : 0.0f;
    dst[(size_t)r * ld + c] = (bf16)v;
}

__global__ void k_fillf(float* __restrict__ p, int n) {
    int i = blockIdx.x * blockDim.x + threadIdx.x;
    if (i < n) p[i] = 0.0f;
}

// ---------------------------------------------------------------------------
// WMMA bf16 GEMM: C[M,N] = A[M,K] * Wt[N,K]^T (+bias) (+relu)
// Block = 256 threads = 8 waves as 4(M)x2(N) over a 128M x 128N tile.
// Wave tile = 32M x 64N = 8 v_wmma_f32_16x16x32_bf16 per K-step.
// B panel (128 cols x 32 K) staged in LDS via async copy, double-buffered;
// A fragments register-pipelined one K-step ahead.
// Requires M % 128 == 0 and N % 16 == 0 (stores masked to n < N).
// ---------------------------------------------------------------------------
#define LDS_CS 40   // LDS column stride in elements (80B: 16B-aligned, conflict-free)

__global__ __launch_bounds__(256)
void k_gemm(const bf16* __restrict__ A, int lda,
            const bf16* __restrict__ Wt,
            const float* __restrict__ bias,
            float* __restrict__ C, int ldc,
            bf16* __restrict__ Cbf, int ldcbf,
            int M, int N, int K, int relu) {
    __shared__ bf16 Bsh[2][128 * LDS_CS];

    const int tid  = threadIdx.x;
    const int wave = tid >> 5;
    const int lane = tid & 31;
    const int wm   = wave >> 1;          // 0..3 -> M sub-tile
    const int wn   = wave & 1;           // 0..1 -> N sub-tile
    const int lh   = lane >> 4;          // half-wave
    const int ll   = lane & 15;

    const int m0 = blockIdx.y * 128 + wm * 32;
    const int n0w = blockIdx.x * 128 + wn * 64;
    const int n0b = blockIdx.x * 128;

    // LDS fill assignment: thread -> (column, 16-element half of the K-slice)
    const int col_f  = tid >> 1;
    const int half_f = tid & 1;
    const bf16* wsrc = Wt + (size_t)(n0b + col_f) * K + half_f * 16;
    bf16* ldst0 = &Bsh[0][col_f * LDS_CS + half_f * 16];
    bf16* ldst1 = &Bsh[1][col_f * LDS_CS + half_f * 16];

    // A fragment pointers (ISA 16-bit A layout: lanes 0-15 K 0-7/16-23,
    // lanes 16-31 K 8-15/24-31).
    const bf16* arow0 = A + (size_t)(m0 + ll) * lda + lh * 8;
    const bf16* arow1 = arow0 + (size_t)16 * lda;

    // Prologue: stage k=0 panel, preload A fragments for k=0.
    fill16(wsrc, ldst0);
    v16bf a0 = load_frag(arow0, arow0 + 16);
    v16bf a1 = load_frag(arow1, arow1 + 16);
    wait_fill();
    __syncthreads();

    v8f acc[8] = {};
    for (int k = 0; k < K; k += 32) {
        const int cur = (k >> 5) & 1;
        const bool more = (k + 32) < K;

        // Issue next stage's async B fill + next A fragment loads first.
        if (more) fill16(wsrc + (k + 32), cur ? ldst0 : ldst1);
        v16bf a0n = a0, a1n = a1;
        if (more) {
            a0n = load_frag(arow0 + k + 32, arow0 + k + 48);
            a1n = load_frag(arow1 + k + 32, arow1 + k + 48);
        }

        // Compute from current LDS stage (ds_load_b128 pairs, bank-conflict-free).
#pragma unroll
        for (int j = 0; j < 4; ++j) {
            const bf16* bp = &Bsh[cur][(wn * 64 + j * 16 + ll) * LDS_CS + lh * 8];
            v16bf bfr = load_frag(bp, bp + 16);
            acc[j]     = __builtin_amdgcn_wmma_f32_16x16x32_bf16(
                false, a0, false, bfr, (short)0, acc[j],     false, false);
            acc[4 + j] = __builtin_amdgcn_wmma_f32_16x16x32_bf16(
                false, a1, false, bfr, (short)0, acc[4 + j], false, false);
        }

        wait_fill();          // this wave's async fills landed
        __syncthreads();      // all waves done reading cur / filling nxt
        a0 = a0n; a1 = a1n;
    }

    // Epilogue. D layout: lane ll = column, VGPR e = row e + lh*8.
#pragma unroll
    for (int r = 0; r < 2; ++r) {
#pragma unroll
        for (int j = 0; j < 4; ++j) {
            const int n = n0w + j * 16 + ll;
            if (n0w + j * 16 >= N) continue;       // mask partial tiles (N=64 case)
            const float bj = bias ? bias[n] : 0.0f;
            const v8f a = acc[r * 4 + j];
#pragma unroll
            for (int e = 0; e < 8; ++e) {
                const int m = m0 + r * 16 + lh * 8 + e;
                float v = a[e] + bj;
                if (relu) v = v > 0.0f ? v : 0.0f;
                if (C)   C[(size_t)m * ldc + n] = v;
                if (Cbf) Cbf[(size_t)m * ldcbf + n] = (bf16)v;
            }
        }
    }
}

// ---------------------------------------------------------------------------
// LSTM gates: z[B,4H] (pre-activations incl. bias, gate order i,f,g,o),
// update c (fp32), write h as bf16 to up to two strided destinations.
// ---------------------------------------------------------------------------
DEVINL float sigm(float x) { return 1.0f / (1.0f + __expf(-x)); }
DEVINL float tanh_f(float x) { return 2.0f / (1.0f + __expf(-2.0f * x)) - 1.0f; }

__global__ void k_gates(const float* __restrict__ z, float* __restrict__ c,
                        bf16* __restrict__ h0, long long ld0,
                        bf16* __restrict__ h1, long long ld1) {
    int i = blockIdx.x * blockDim.x + threadIdx.x;   // exactly B*H threads
    int r   = i >> 10;                               // / H   (H = 1024)
    int col = i & 1023;                              // % H
    const float* zr = z + (size_t)r * 4096;
    float ig = sigm(zr[col]);
    float fg = sigm(zr[1024 + col]);
    float gg = tanh_f(zr[2048 + col]);
    float og = sigm(zr[3072 + col]);
    float cn = fg * c[i] + ig * gg;
    c[i] = cn;
    float h = og * tanh_f(cn);
    h0[(size_t)r * ld0 + col] = (bf16)h;
    if (h1) h1[(size_t)r * ld1 + col] = (bf16)h;
}

// ---------------------------------------------------------------------------
// Attention: grid = B blocks of 128 threads (wave = head).
// q fp32 [B,H]; Kp,Vp bf16 [B,S,NH,HD]; ctx bf16 [B,H].
// ---------------------------------------------------------------------------
__global__ __launch_bounds__(128)
void k_attn(const float* __restrict__ q, const bf16* __restrict__ Kp,
            const bf16* __restrict__ Vp, bf16* __restrict__ ctx) {
    __shared__ float sw[4 * 128];
    const int b    = blockIdx.x;
    const int head = threadIdx.x >> 5;
    const int lane = threadIdx.x & 31;

    const float* qh  = q  + (size_t)b * 1024 + head * 256;
    const bf16*  KpB = Kp + (size_t)b * 128 * 1024 + head * 256;
    const bf16*  VpB = Vp + (size_t)b * 128 * 1024 + head * 256;

    // scores: lane handles s = lane, lane+32, lane+64, lane+96
    float sc[4] = {0.f, 0.f, 0.f, 0.f};
    for (int d = 0; d < 256; d += 8) {
        float qv[8];
#pragma unroll
        for (int j = 0; j < 8; ++j) qv[j] = qh[d + j];
#pragma unroll
        for (int i = 0; i < 4; ++i) {
            const int s = lane + i * 32;
            v8bf kv = *(const v8bf*)(KpB + (size_t)s * 1024 + d);
#pragma unroll
            for (int j = 0; j < 8; ++j) sc[i] += qv[j] * (float)kv[j];
        }
    }
#pragma unroll
    for (int i = 0; i < 4; ++i) sc[i] *= (1.0f / 16.0f);   // /sqrt(HD)

    // softmax over S=128 (4 per lane + wave32 shuffle reduce)
    float m = fmaxf(fmaxf(sc[0], sc[1]), fmaxf(sc[2], sc[3]));
#pragma unroll
    for (int off = 16; off > 0; off >>= 1) m = fmaxf(m, __shfl_xor(m, off, 32));
    float e[4], sum = 0.0f;
#pragma unroll
    for (int i = 0; i < 4; ++i) { e[i] = __expf(sc[i] - m); sum += e[i]; }
#pragma unroll
    for (int off = 16; off > 0; off >>= 1) sum += __shfl_xor(sum, off, 32);
    const float inv = 1.0f / sum;
#pragma unroll
    for (int i = 0; i < 4; ++i) sw[head * 128 + lane + i * 32] = e[i] * inv;
    __syncthreads();

    // ctx[d] = sum_s w_s * V[s,d]; lane owns 8 consecutive d.
    float acc8[8] = {};
    const bf16* vp0 = VpB + lane * 8;
    for (int s = 0; s < 128; ++s) {
        float w = sw[head * 128 + s];
        v8bf vv = *(const v8bf*)(vp0 + (size_t)s * 1024);
#pragma unroll
        for (int j = 0; j < 8; ++j) acc8[j] += w * (float)vv[j];
    }
    bf16* cd = ctx + (size_t)b * 1024 + head * 256 + lane * 8;
#pragma unroll
    for (int j = 0; j < 8; ++j) cd[j] = (bf16)acc8[j];
}

// ---------------------------------------------------------------------------
// Host orchestration
// ---------------------------------------------------------------------------
extern "C" void kernel_launch(void* const* d_in, const int* in_sizes, int n_in,
                              void* d_out, int out_size, void* d_ws, size_t ws_size,
                              hipStream_t stream) {
    (void)in_sizes; (void)n_in; (void)out_size; (void)ws_size;
    constexpr int B = 256, S = 128, IN = 64, OUT = 64, H = 1024, STEPS = 16;

    const float* x     = (const float*)d_in[0];
    const float* e0_Wx = (const float*)d_in[1];
    const float* e0_Wh = (const float*)d_in[2];
    const float* e0_b  = (const float*)d_in[3];
    const float* e1_Wx = (const float*)d_in[4];
    const float* e1_Wh = (const float*)d_in[5];
    const float* e1_b  = (const float*)d_in[6];
    const float* d0_Wx = (const float*)d_in[7];
    const float* d0_Wh = (const float*)d_in[8];
    const float* d0_b  = (const float*)d_in[9];
    const float* d1_Wx = (const float*)d_in[10];
    const float* d1_Wh = (const float*)d_in[11];
    const float* d1_b  = (const float*)d_in[12];
    const float* Wq    = (const float*)d_in[13];
    const float* Wk    = (const float*)d_in[14];
    const float* Wv    = (const float*)d_in[15];
    const float* Wo    = (const float*)d_in[16];
    const float* bq    = (const float*)d_in[17];
    const float* bk    = (const float*)d_in[18];
    const float* bv    = (const float*)d_in[19];
    const float* bo    = (const float*)d_in[20];
    const float* fc_W  = (const float*)d_in[21];
    const float* fc_b  = (const float*)d_in[22];
    const float* out_W = (const float*)d_in[23];
    const float* out_b = (const float*)d_in[24];
    float* out = (float*)d_out;

    // --- workspace carve (256B aligned) ---
    char* p = (char*)d_ws;
    auto carve = [&](size_t bytes) -> char* {
        char* r = p; p += (bytes + 255) & ~(size_t)255; return r;
    };
    bf16* Wt_e0  = (bf16*)carve((size_t)4096 * 1088 * 2);
    bf16* Wt_e1  = (bf16*)carve((size_t)4096 * 2048 * 2);
    bf16* Wt_d0  = (bf16*)carve((size_t)4096 * 1088 * 2);
    bf16* Wt_d1  = (bf16*)carve((size_t)4096 * 2048 * 2);
    bf16* Wq_t   = (bf16*)carve((size_t)1024 * 1024 * 2);
    bf16* Wk_t   = (bf16*)carve((size_t)1024 * 1024 * 2);
    bf16* Wv_t   = (bf16*)carve((size_t)1024 * 1024 * 2);
    bf16* Wo_t   = (bf16*)carve((size_t)1024 * 1024 * 2);
    bf16* fc_Wt  = (bf16*)carve((size_t)1024 * 2048 * 2);
    bf16* out_Wt = (bf16*)carve((size_t)128 * 1024 * 2);   // 64 cols + fill slack
    bf16* A0     = (bf16*)carve((size_t)B * 1088 * 2);     // [x_t | h0]
    bf16* A1     = (bf16*)carve((size_t)B * 2048 * 2);     // [y0_t | h1]
    float* zbuf  = (float*)carve((size_t)B * 4096 * 4);
    float* c0    = (float*)carve((size_t)B * H * 4);
    float* c1    = (float*)carve((size_t)B * H * 4);
    bf16* enc    = (bf16*)carve((size_t)B * S * H * 2);    // encoder outputs
    bf16* Kp     = (bf16*)carve((size_t)B * S * H * 2);
    bf16* Vp     = (bf16*)carve((size_t)B * S * H * 2);
    float* qbuf  = (float*)carve((size_t)B * H * 4);
    bf16* ctx    = (bf16*)carve((size_t)B * H * 2);
    bf16* fcin   = (bf16*)carve((size_t)B * 2048 * 2);     // [dh1 | attn]
    bf16* comb   = (bf16*)carve((size_t)B * H * 2);

    // --- launch helpers ---
    auto wconv = [&](const float* wx, int kx, const float* wh, int kh,
                     bf16* wt, int n) {
        long long tot = (long long)n * (kx + kh);
        k_wconv<<<(unsigned)((tot + 255) / 256), 256, 0, stream>>>(wx, kx, wh, kh, wt, n);
    };
    auto cvt = [&](const float* src, long long ls, bf16* dst, long long ld,
                   int rows, int cols) {
        int tot = rows * cols;
        k_cvt<<<(tot + 255) / 256, 256, 0, stream>>>(src, ls, dst, ld, rows, cols);
    };
    auto gemm = [&](const bf16* A, int lda, const bf16* Wt, const float* bias,
                    float* C, int ldc, bf16* Cbf, int ldcbf,
                    int M, int N, int K, int relu) {
        dim3 g((N + 127) / 128, M / 128);
        k_gemm<<<g, 256, 0, stream>>>(A, lda, Wt, bias, C, ldc, Cbf, ldcbf,
                                      M, N, K, relu);
    };
    auto gates = [&](float* z, float* c, bf16* h0, long long ld0,
                     bf16* h1, long long ld1) {
        k_gates<<<(B * H) / 256, 256, 0, stream>>>(z, c, h0, ld0, h1, ld1);
    };

    // --- weight prep (once per call) ---
    wconv(e0_Wx, IN, e0_Wh, H, Wt_e0, 4096);
    wconv(e1_Wx, H,  e1_Wh, H, Wt_e1, 4096);
    wconv(d0_Wx, IN, d0_Wh, H, Wt_d0, 4096);
    wconv(d1_Wx, H,  d1_Wh, H, Wt_d1, 4096);
    wconv(Wq, H, nullptr, 0, Wq_t, H);
    wconv(Wk, H, nullptr, 0, Wk_t, H);
    wconv(Wv, H, nullptr, 0, Wv_t, H);
    wconv(Wo, H, nullptr, 0, Wo_t, H);
    wconv(fc_W, 2 * H, nullptr, 0, fc_Wt, H);
    wconv(out_W, H, nullptr, 0, out_Wt, OUT);

    // --- state init (must be re-done every call: harness does not re-poison)
    cvt(nullptr, 0, A0, 1088, B, 1088);
    cvt(nullptr, 0, A1, 2048, B, 2048);
    k_fillf<<<(B * H + 255) / 256, 256, 0, stream>>>(c0, B * H);
    k_fillf<<<(B * H + 255) / 256, 256, 0, stream>>>(c1, B * H);

    // --- encoder: both layers fused per timestep ---
    for (int t = 0; t < S; ++t) {
        cvt(x + (size_t)t * IN, (long long)S * IN, A0, 1088, B, IN);  // x_t
        gemm(A0, 1088, Wt_e0, e0_b, zbuf, 4096, nullptr, 0, B, 4096, 1088, 0);
        gates(zbuf, c0, A0 + 64, 1088, A1, 2048);        // h0 -> recur + e1 in
        gemm(A1, 2048, Wt_e1, e1_b, zbuf, 4096, nullptr, 0, B, 4096, 2048, 0);
        gates(zbuf, c1, A1 + 1024, 2048, enc + (size_t)t * H, (long long)S * H);
    }

    // --- K/V projections over full encoder output (big parallel GEMMs) ---
    gemm(enc, H, Wk_t, bk, nullptr, 0, Kp, H, B * S, H, H, 0);
    gemm(enc, H, Wv_t, bv, nullptr, 0, Vp, H, B * S, H, H, 0);

    // --- decoder: dh0/dc0/dh1/dc1 carry over from encoder state in A0/A1/c ---
    cvt(nullptr, 0, A0, 1088, B, IN);                    // xin = zeros
    for (int sd = 0; sd < STEPS; ++sd) {
        gemm(A0, 1088, Wt_d0, d0_b, zbuf, 4096, nullptr, 0, B, 4096, 1088, 0);
        gates(zbuf, c0, A0 + 64, 1088, A1, 2048);        // dh0 -> recur + d1 in
        gemm(A1, 2048, Wt_d1, d1_b, zbuf, 4096, nullptr, 0, B, 4096, 2048, 0);
        gates(zbuf, c1, A1 + 1024, 2048, fcin, 2048);    // dh1 -> recur + fc in
        gemm(A1 + 1024, 2048, Wq_t, bq, qbuf, 1024, nullptr, 0, B, H, H, 0);
        k_attn<<<B, 128, 0, stream>>>(qbuf, Kp, Vp, ctx);
        gemm(ctx, H, Wo_t, bo, nullptr, 0, fcin + 1024, 2048, B, H, H, 0);
        gemm(fcin, 2048, fc_Wt, fc_b, nullptr, 0, comb, H, B, H, 2 * H, 1);
        // out -> d_out[:, sd, :] (fp32) and next xin (bf16 into A0[:, :64])
        gemm(comb, H, out_Wt, out_b, out + (size_t)sd * OUT, STEPS * OUT,
             A0, 1088, B, OUT, H, 0);
    }
}
